// MyDeepAR_171798692054
// MI455X (gfx1250) — matrix operand order repaired and checked by reading, not compiled, not run on hardware
//
#include <hip/hip_runtime.h>
#include <hip/hip_bf16.h>

typedef __attribute__((ext_vector_type(16))) __bf16 v16bf;
typedef __attribute__((ext_vector_type(8)))  float  v8f;

#define NSAMP   200
#define BATCH   128
#define TSTEPS  32
#define HIDN    64
#define ROWS    128      // rows of the 25600-row virtual batch per block
#define NTHREADS 256     // 8 waves (wave32), one 16-row M tile per wave

// ---- LDS layout (ushort region then float region) ----
#define OFF_WIH0 0                           // [256][32] bf16
#define OFF_WHH0 (OFF_WIH0 + 256*32)         // [256][64] bf16
#define OFF_WIH1 (OFF_WHH0 + 256*64)         // [256][64] bf16
#define OFF_WHH1 (OFF_WIH1 + 256*64)         // [256][64] bf16
#define OFF_H    (OFF_WHH1 + 256*64)         // [2][128][64] bf16 hidden state
#define OFF_XB   (OFF_H + 2*ROWS*HIDN)       // [2][32] bf16 per-b feature vec
#define US_COUNT (OFF_XB + 64)
#define FBYTE_OFF (US_COUNT*2)
#define L_LAG 0          // [128] f32 lagged target
#define L_WPJ 128        // [2][64] f32 projection weights
#define L_BPJ 256        // [2]  f32 projection bias
#define SMEM_BYTES (FBYTE_OFF + 258*4 + 32)

__device__ __forceinline__ unsigned short f2bf(float f) {
    unsigned int x = __builtin_bit_cast(unsigned int, f);
    x += 0x7FFFu + ((x >> 16) & 1u);               // round-to-nearest-even
    return (unsigned short)(x >> 16);
}
__device__ __forceinline__ float bf2f(unsigned short h) {
    unsigned int x = ((unsigned int)h) << 16;
    return __builtin_bit_cast(float, x);
}
__device__ __forceinline__ float sigm(float x) { return 1.0f / (1.0f + __expf(-x)); }
__device__ __forceinline__ float tanh_fast(float x) { return 2.0f * sigm(2.0f * x) - 1.0f; }

union ABFrag { v16bf v; unsigned int u[8]; };

// A fragment (16x32 bf16): lane covers row lane%16; K = {8h..8h+7} and {16+8h..16+8h+7}
__device__ __forceinline__ v16bf load_a(const unsigned short* base, int row, int kc, int half) {
    ABFrag f;
    const unsigned int* p0 = (const unsigned int*)(base + row * HIDN + kc * 32 + 8 * half);
    const unsigned int* p1 = (const unsigned int*)(base + row * HIDN + kc * 32 + 16 + 8 * half);
    f.u[0] = p0[0]; f.u[1] = p0[1]; f.u[2] = p0[2]; f.u[3] = p0[3];
    f.u[4] = p1[0]; f.u[5] = p1[1]; f.u[6] = p1[2]; f.u[7] = p1[3];
    return f.v;
}
// B fragment (32x16 bf16): lane covers column lane%16 (= W row n); K = {8h..8h+7} and {16+8h..16+8h+7}
__device__ __forceinline__ v16bf load_b(const unsigned short* w, int stride, int n, int kBase, int half) {
    ABFrag f;
    const unsigned int* p = (const unsigned int*)(w + n * stride + kBase + 8 * half);
    const unsigned int* q = (const unsigned int*)(w + n * stride + kBase + 16 + 8 * half);
#pragma unroll
    for (int i = 0; i < 4; ++i) { f.u[i] = p[i]; f.u[4 + i] = q[i]; }
    return f.v;
}

#define WMMA_BF16(a, b, c) \
    __builtin_amdgcn_wmma_f32_16x16x32_bf16(false, (a), false, (b), (short)0, (c), false, false)

__device__ __forceinline__ float sample_normal(unsigned int r, unsigned int t) {
    unsigned int s = r * 0x9E3779B9u + t * 0x85EBCA6Bu + 0x6A09E667u;
    s ^= s >> 16; s *= 0x7FEB352Du; s ^= s >> 15; s *= 0x846CA68Bu; s ^= s >> 16;
    unsigned int q = s + 0x9E3779B9u;
    q ^= q >> 16; q *= 0x7FEB352Du; q ^= q >> 15; q *= 0x846CA68Bu; q ^= q >> 16;
    float u1 = fmaxf((float)(s >> 8) * (1.0f / 16777216.0f), 1.0e-7f);
    float u2 = (float)(q >> 8) * (1.0f / 16777216.0f);
    float rad = __fsqrt_rn(-2.0f * __logf(u1));
    return rad * __cosf(6.28318530718f * u2);
}

__global__ __launch_bounds__(NTHREADS, 1)
void deepar_persistent(const float* __restrict__ dcont,  const float* __restrict__ oneoff,
                       const float* __restrict__ tscale, const float* __restrict__ emb0,
                       const float* __restrict__ emb1,   const float* __restrict__ Wih0,
                       const float* __restrict__ Whh0,   const float* __restrict__ bih0,
                       const float* __restrict__ bhh0,   const float* __restrict__ Wih1,
                       const float* __restrict__ Whh1,   const float* __restrict__ bih1,
                       const float* __restrict__ bhh1,   const float* __restrict__ Wproj,
                       const float* __restrict__ bproj,  const float* __restrict__ h0,
                       const float* __restrict__ c0,     const int* __restrict__ dcat,
                       float* __restrict__ out) {
    extern __shared__ char smem[];
    unsigned short* sU = (unsigned short*)smem;
    float* sF = (float*)(smem + FBYTE_OFF);

    const int tid   = threadIdx.x;
    const int lane  = tid & 31;
    const int wave  = tid >> 5;
    const int ln16  = lane & 15;
    const int half  = lane >> 4;
    const int wrow0 = wave * 16;
    const int r0    = blockIdx.x * ROWS;
    const int bBase = r0 / NSAMP;

    // ---- setup: weights -> LDS bf16 (one time; reused across all 32 steps) ----
    for (int i = tid; i < 256 * 32; i += NTHREADS) sU[OFF_WIH0 + i] = f2bf(Wih0[i]);
    for (int i = tid; i < 256 * 64; i += NTHREADS) sU[OFF_WHH0 + i] = f2bf(Whh0[i]);
    for (int i = tid; i < 256 * 64; i += NTHREADS) sU[OFF_WIH1 + i] = f2bf(Wih1[i]);
    for (int i = tid; i < 256 * 64; i += NTHREADS) sU[OFF_WHH1 + i] = f2bf(Whh1[i]);
    // hidden state h (sample-expanded) -> LDS bf16
    for (int i = tid; i < 2 * ROWS * HIDN; i += NTHREADS) {
        int l = i >> 13, rem = i & 8191, r = rem >> 6, j = rem & 63;
        int gb = (r0 + r) / NSAMP;
        sU[OFF_H + i] = f2bf(h0[l * BATCH * HIDN + gb * HIDN + j]);
    }
    for (int i = tid; i < 128; i += NTHREADS) sF[L_WPJ + i] = Wproj[i];
    if (tid < 2) sF[L_BPJ + tid] = bproj[tid];
    if (tid < ROWS) sF[L_LAG + tid] = oneoff[(r0 + tid) / NSAMP];   // first_target

    // per-lane gate biases (column = lane%16 within each 16-wide hid tile)
    float bias0[16], bias1[16];
#pragma unroll
    for (int t = 0; t < 4; ++t)
#pragma unroll
        for (int g = 0; g < 4; ++g) {
            int idx = g * 64 + t * 16 + ln16;
            bias0[t * 4 + g] = bih0[idx] + bhh0[idx];
            bias1[t * 4 + g] = bih1[idx] + bhh1[idx];
        }

    // cell state in registers, C/D-matrix layout: row = wrow0 + r + 8*half, col = t*16+ln16
    float c0r[32], c1r[32];
#pragma unroll
    for (int t = 0; t < 4; ++t)
#pragma unroll
        for (int r = 0; r < 8; ++r) {
            int R  = r0 + wrow0 + r + 8 * half;
            int gb = R / NSAMP;
            int j  = t * 16 + ln16;
            c0r[t * 8 + r] = c0[0 * BATCH * HIDN + gb * HIDN + j];
            c1r[t * 8 + r] = c0[1 * BATCH * HIDN + gb * HIDN + j];
        }

    float myCenter = 0.0f, myScale = 1.0f;
    if (tid < ROWS) {
        int gb = (r0 + tid) / NSAMP;
        myCenter = tscale[gb * 2 + 0];
        myScale  = tscale[gb * 2 + 1];
    }
    __syncthreads();

    for (int ts = 0; ts < TSTEPS; ++ts) {
        // ---- build per-b feature vector (block spans <=2 distinct b) ----
        if (tid < 64) {
            int vb = tid >> 5, k = tid & 31;
            int bLast = (r0 + ROWS - 1) / NSAMP;
            int bq = bBase + vb; if (bq > bLast) bq = bLast;
            float v = 0.0f;
            if (k >= 1 && k < 6)       v = dcont[bq * TSTEPS * 6 + ts * 6 + k];
            else if (k >= 6 && k < 16) { int c = dcat[bq * TSTEPS * 2 + ts * 2 + 0]; v = emb0[c * 10 + (k - 6)]; }
            else if (k >= 16)          { int c = dcat[bq * TSTEPS * 2 + ts * 2 + 1]; v = emb1[c * 16 + (k - 16)]; }
            sU[OFF_XB + vb * 32 + k] = f2bf(v);
        }
        __syncthreads();

        // ---- layer 0: A fragments (x_t with lagged target spliced at K=0, plus h0 state) ----
        ABFrag fx;
        {
            int xr = wrow0 + ln16;
            int vb = (r0 + xr) / NSAMP - bBase;
            const unsigned int* p0 = (const unsigned int*)(sU + OFF_XB + vb * 32 + 8 * half);
            const unsigned int* p1 = (const unsigned int*)(sU + OFF_XB + vb * 32 + 16 + 8 * half);
            fx.u[0] = p0[0]; fx.u[1] = p0[1]; fx.u[2] = p0[2]; fx.u[3] = p0[3];
            fx.u[4] = p1[0]; fx.u[5] = p1[1]; fx.u[6] = p1[2]; fx.u[7] = p1[3];
            if (half == 0)
                fx.u[0] = (fx.u[0] & 0xFFFF0000u) | (unsigned int)f2bf(sF[L_LAG + xr]);
        }
        v16bf a_h00 = load_a(sU + OFF_H, wrow0 + ln16, 0, half);
        v16bf a_h01 = load_a(sU + OFF_H, wrow0 + ln16, 1, half);

#pragma unroll
        for (int t = 0; t < 4; ++t) {
            v8f acc[4];
#pragma unroll
            for (int g = 0; g < 4; ++g) {
                float bv = bias0[t * 4 + g];
                v8f c = {bv, bv, bv, bv, bv, bv, bv, bv};
                int n = g * 64 + t * 16 + ln16;
                c = WMMA_BF16(fx.v,  load_b(sU + OFF_WIH0, 32, n, 0,  half), c);
                c = WMMA_BF16(a_h00, load_b(sU + OFF_WHH0, 64, n, 0,  half), c);
                c = WMMA_BF16(a_h01, load_b(sU + OFF_WHH0, 64, n, 32, half), c);
                acc[g] = c;
            }
#pragma unroll
            for (int r = 0; r < 8; ++r) {
                float iv = acc[0][r], fv = acc[1][r], gv = acc[2][r], ov = acc[3][r];
                float cc = c0r[t * 8 + r];
                cc = sigm(fv) * cc + sigm(iv) * tanh_fast(gv);
                float hh = sigm(ov) * tanh_fast(cc);
                c0r[t * 8 + r] = cc;
                sU[OFF_H + (wrow0 + r + 8 * half) * HIDN + t * 16 + ln16] = f2bf(hh);
            }
        }
        __syncthreads();

        // ---- layer 1: input = new h0, recurrent = old h1 (fragments loaded before overwrite) ----
        v16bf a_i0  = load_a(sU + OFF_H, wrow0 + ln16, 0, half);
        v16bf a_i1  = load_a(sU + OFF_H, wrow0 + ln16, 1, half);
        v16bf a_h10 = load_a(sU + OFF_H + ROWS * HIDN, wrow0 + ln16, 0, half);
        v16bf a_h11 = load_a(sU + OFF_H + ROWS * HIDN, wrow0 + ln16, 1, half);

#pragma unroll
        for (int t = 0; t < 4; ++t) {
            v8f acc[4];
#pragma unroll
            for (int g = 0; g < 4; ++g) {
                float bv = bias1[t * 4 + g];
                v8f c = {bv, bv, bv, bv, bv, bv, bv, bv};
                int n = g * 64 + t * 16 + ln16;
                c = WMMA_BF16(a_i0,  load_b(sU + OFF_WIH1, 64, n, 0,  half), c);
                c = WMMA_BF16(a_i1,  load_b(sU + OFF_WIH1, 64, n, 32, half), c);
                c = WMMA_BF16(a_h10, load_b(sU + OFF_WHH1, 64, n, 0,  half), c);
                c = WMMA_BF16(a_h11, load_b(sU + OFF_WHH1, 64, n, 32, half), c);
                acc[g] = c;
            }
#pragma unroll
            for (int r = 0; r < 8; ++r) {
                float iv = acc[0][r], fv = acc[1][r], gv = acc[2][r], ov = acc[3][r];
                float cc = c1r[t * 8 + r];
                cc = sigm(fv) * cc + sigm(iv) * tanh_fast(gv);
                float hh = sigm(ov) * tanh_fast(cc);
                c1r[t * 8 + r] = cc;
                sU[OFF_H + ROWS * HIDN + (wrow0 + r + 8 * half) * HIDN + t * 16 + ln16] = f2bf(hh);
            }
        }
        __syncthreads();

        // ---- projection (2x64) + sampling + lagged-target feedback ----
        if (tid < ROWS) {
            const unsigned short* hrow = sU + OFF_H + ROWS * HIDN + tid * HIDN;
            float d0 = sF[L_BPJ + 0], d1 = sF[L_BPJ + 1];
#pragma unroll 8
            for (int j = 0; j < HIDN; ++j) {
                float hv = bf2f(hrow[j]);
                d0 += hv * sF[L_WPJ + j];
                d1 += hv * sF[L_WPJ + 64 + j];
            }
            float loc   = d0 * myScale + myCenter;
            float sp    = (d1 > 20.0f) ? d1 : __logf(1.0f + __expf(d1));
            float sigma = sp * myScale;
            int R = r0 + tid;
            float pred = loc + sigma * sample_normal((unsigned)R, (unsigned)ts);
            int gb = R / NSAMP, s = R - gb * NSAMP;
            out[gb * (TSTEPS * NSAMP) + ts * NSAMP + s] = pred;
            sF[L_LAG + tid] = (pred - myCenter) / myScale;
        }
        // lag/xb hazards are closed by the post-xb-build __syncthreads of the next iteration
    }
}

extern "C" void kernel_launch(void* const* d_in, const int* in_sizes, int n_in,
                              void* d_out, int out_size, void* d_ws, size_t ws_size,
                              hipStream_t stream) {
    (void)in_sizes; (void)n_in; (void)out_size; (void)d_ws; (void)ws_size;
    const float* dcont  = (const float*)d_in[0];
    const float* oneoff = (const float*)d_in[1];
    const float* tscale = (const float*)d_in[2];
    const float* emb0   = (const float*)d_in[3];
    const float* emb1   = (const float*)d_in[4];
    const float* Wih0   = (const float*)d_in[5];
    const float* Whh0   = (const float*)d_in[6];
    const float* bih0   = (const float*)d_in[7];
    const float* bhh0   = (const float*)d_in[8];
    const float* Wih1   = (const float*)d_in[9];
    const float* Whh1   = (const float*)d_in[10];
    const float* bih1   = (const float*)d_in[11];
    const float* bhh1   = (const float*)d_in[12];
    const float* Wproj  = (const float*)d_in[13];
    const float* bproj  = (const float*)d_in[14];
    const float* h0     = (const float*)d_in[15];
    const float* c0     = (const float*)d_in[16];
    const int*   dcat   = (const int*)d_in[17];

    hipFuncSetAttribute((const void*)deepar_persistent,
                        hipFuncAttributeMaxDynamicSharedMemorySize, SMEM_BYTES);
    dim3 grid((BATCH * NSAMP) / ROWS);   // 200 persistent blocks
    deepar_persistent<<<grid, NTHREADS, SMEM_BYTES, stream>>>(
        dcont, oneoff, tscale, emb0, emb1, Wih0, Whh0, bih0, bhh0,
        Wih1, Whh1, bih1, bhh1, Wproj, bproj, h0, c0, dcat, (float*)d_out);
}
